// AttentionReadout_48679159333012
// MI455X (gfx1250) — matrix operand order repaired and checked by reading, not compiled, not run on hardware
//
#include <hip/hip_runtime.h>
#include <hip/hip_bf16.h>
#include <math.h>

#define B_    4
#define S_    1024
#define E_    768
#define H_    12
#define HD_   64
#define KV_   2049
#define E2_   1536
#define E4_   3072
#define MLP_  3072
#define SOUT_ 256
#define L_    2

typedef __attribute__((ext_vector_type(16))) __bf16 v16bf;
typedef __attribute__((ext_vector_type(8)))  float  v8f;

union FragBF { v16bf v; unsigned u[8]; };

__device__ __forceinline__ unsigned short f2bf(float f) {
  union { float f; unsigned u; } c; c.f = f;
  return (unsigned short)((c.u + 0x7FFFu + ((c.u >> 16) & 1u)) >> 16);
}

__device__ __forceinline__ float gelu_f(float x) {
  return 0.5f * x * (1.0f + erff(x * 0.70710678118654752f));
}

#define WMMA_BF16(a, b, c) \
  __builtin_amdgcn_wmma_f32_16x16x32_bf16(false, (a), false, (b), (short)0, (c), false, false)

// ---------------------------------------------------------------------------
// gfx1250 async global->LDS path (ASYNCcnt-tracked), feature-guarded so the
// file still compiles (fallback: load + ds_store) if the builtins are absent.
// ---------------------------------------------------------------------------
#if defined(__has_builtin)
#  if __has_builtin(__builtin_amdgcn_global_load_async_to_lds_b128) && \
      __has_builtin(__builtin_amdgcn_s_wait_asynccnt)
#    define USE_ASYNC_LDS 1
#  endif
#endif

#ifdef USE_ASYNC_LDS
typedef int v4i __attribute__((vector_size(16)));
__device__ __forceinline__ void g2l_async16(const unsigned short* g, unsigned short* l) {
  __builtin_amdgcn_global_load_async_to_lds_b128(
      (__attribute__((address_space(1))) v4i*)(size_t)g,
      (__attribute__((address_space(3))) v4i*)(unsigned)(size_t)l,
      0, 0);
}
#endif

// ---------------------------------------------------------------------------
// fp32 -> bf16 conversion (weights)
// ---------------------------------------------------------------------------
__global__ __launch_bounds__(256) void f2bf_kernel(const float* __restrict__ in,
                                                   unsigned short* __restrict__ out, int n) {
  for (int i = blockIdx.x * 256 + threadIdx.x; i < n; i += gridDim.x * 256)
    out[i] = f2bf(in[i]);
}

// ---------------------------------------------------------------------------
// temb row: bf16 copy into A-matrix buffer and into all_embed row 0
// ---------------------------------------------------------------------------
__global__ __launch_bounds__(256) void temb_prep_kernel(const float* __restrict__ temb,
                                                        unsigned short* __restrict__ tembb,
                                                        unsigned short* __restrict__ allemb) {
  int b = blockIdx.x;
  for (int e = threadIdx.x; e < E_; e += 256) {
    unsigned short h = f2bf(temb[(size_t)b * E_ + e]);
    tembb[(size_t)b * E_ + e] = h;
    allemb[((size_t)b * KV_) * E_ + e] = h;  // row 0 of batch b (raw temb, no LN)
  }
}

// ---------------------------------------------------------------------------
// Prep: inputs = l2r + r2l; layernorm both streams; qin = ln1+ln2 (bf16);
// all_embed rows 1..S = ln1, rows S+1..2S = ln2 (bf16)
// ---------------------------------------------------------------------------
__global__ __launch_bounds__(256) void prep_kernel(
    const float* __restrict__ l2r, const float* __restrict__ r2l,
    const float* __restrict__ g1, const float* __restrict__ b1,
    const float* __restrict__ g2, const float* __restrict__ b2,
    float* __restrict__ inputs_f, unsigned short* __restrict__ qin_b,
    unsigned short* __restrict__ allemb_b) {
  int rowid = blockIdx.x;
  int b = rowid / S_, s = rowid % S_;
  int tid = threadIdx.x;
  const float* pa = l2r + (size_t)rowid * E_;
  const float* pb = r2l + (size_t)rowid * E_;

  float s1 = 0.f, q1 = 0.f, s2 = 0.f, q2 = 0.f;
  for (int e = tid; e < E_; e += 256) {
    float a = pa[e], c = pb[e];
    s1 += a; q1 += a * a; s2 += c; q2 += c * c;
  }
  __shared__ float sh[512];
  sh[tid] = s1; sh[256 + tid] = q1;
  __syncthreads();
  for (int o = 128; o > 0; o >>= 1) {
    if (tid < o) { sh[tid] += sh[tid + o]; sh[256 + tid] += sh[256 + tid + o]; }
    __syncthreads();
  }
  float m1 = sh[0] / (float)E_;
  float v1 = sh[256] / (float)E_ - m1 * m1;
  __syncthreads();
  sh[tid] = s2; sh[256 + tid] = q2;
  __syncthreads();
  for (int o = 128; o > 0; o >>= 1) {
    if (tid < o) { sh[tid] += sh[tid + o]; sh[256 + tid] += sh[256 + tid + o]; }
    __syncthreads();
  }
  float m2 = sh[0] / (float)E_;
  float v2 = sh[256] / (float)E_ - m2 * m2;
  float r1 = rsqrtf(v1 + 1e-5f), r2 = rsqrtf(v2 + 1e-5f);

  for (int e = tid; e < E_; e += 256) {
    float a = pa[e], c = pb[e];
    float n1 = (a - m1) * r1 * g1[e] + b1[e];
    float n2 = (c - m2) * r2 * g2[e] + b2[e];
    inputs_f[(size_t)rowid * E_ + e] = a + c;
    qin_b[(size_t)rowid * E_ + e] = f2bf(n1 + n2);
    allemb_b[((size_t)b * KV_ + 1 + s) * E_ + e] = f2bf(n1);
    allemb_b[((size_t)b * KV_ + 1 + S_ + s) * E_ + e] = f2bf(n2);
  }
}

// ---------------------------------------------------------------------------
// bf16 WMMA GEMM:  C[M,N] = act(A[M,K] @ W[K,N] * scale + bias) + resid
// Block tile 128x128, 256 threads = 8 waves, each wave a 32(M) x 64(N) strip
// -> 8 v_wmma per K-step per wave. Double-buffered LDS; padded stride 40
// halves (80B) => conflict-free fragment loads and 16B-aligned b128 staging.
// A tiles staged via global_load_async_to_lds_b128 (ASYNCcnt) when in-bounds.
// Requires N % 128 == 0, K % 32 == 0 (true for all call sites). M checked.
// ---------------------------------------------------------------------------
__global__ __launch_bounds__(256) void gemm_bf16_kernel(
    const unsigned short* __restrict__ A, const unsigned short* __restrict__ W,
    const float* __restrict__ bias, const float* __restrict__ resid,
    float* __restrict__ Cf, unsigned short* __restrict__ Cb,
    int M, int N, int K, float scale, int act) {
  constexpr int STR = 40;  // padded LDS row stride in halves
  __shared__ __align__(16) unsigned short sA[2][128 * STR];
  __shared__ __align__(16) unsigned short sB[2][128 * STR];  // transposed: [n][k]

  const int tid = threadIdx.x;
  const int m0 = blockIdx.y * 128, n0 = blockIdx.x * 128;
  const int wave = tid >> 5, lane = tid & 31;
  const int wm = (wave & 3) * 32, wn = (wave >> 2) * 64;
  const int col = lane & 15, g = lane >> 4;

  const int a_r = tid >> 1;        // 0..127 : A row in tile
  const int a_c = (tid & 1) * 16;  // 0/16   : A col base (16 halves per thread)
  const int b_k = tid >> 3;        // 0..31  : W k-row in slab
  const int b_n = (tid & 7) * 16;  // 0..112 : W col base (16 halves per thread)

  const bool mfull = (m0 + 128 <= M);

  v8f acc[8];
#pragma unroll
  for (int i = 0; i < 8; ++i) acc[i] = (v8f){0.f, 0.f, 0.f, 0.f, 0.f, 0.f, 0.f, 0.f};

  auto stage = [&](int k0, int buf) {
    // A tile (128 x 32)
#ifdef USE_ASYNC_LDS
    if (mfull) {
      const unsigned short* gp = &A[(size_t)(m0 + a_r) * K + k0 + a_c];
      g2l_async16(gp, &sA[buf][a_r * STR + a_c]);
      g2l_async16(gp + 8, &sA[buf][a_r * STR + a_c + 8]);
    } else
#endif
    {
      uint4 v0 = {0u, 0u, 0u, 0u}, v1 = {0u, 0u, 0u, 0u};
      if (m0 + a_r < M) {
        const unsigned short* gp = &A[(size_t)(m0 + a_r) * K + k0 + a_c];
        v0 = *(const uint4*)gp;
        v1 = *(const uint4*)(gp + 8);
      }
      *(uint4*)&sA[buf][a_r * STR + a_c] = v0;
      *(uint4*)&sA[buf][a_r * STR + a_c + 8] = v1;
    }
    // W tile (32 x 128) stored transposed sB[n][k]
    const unsigned short* wp = &W[(size_t)(k0 + b_k) * N + n0 + b_n];
    uint4 w0 = *(const uint4*)wp;
    uint4 w1 = *(const uint4*)(wp + 8);
    unsigned short t16[16];
    *(uint4*)&t16[0] = w0;
    *(uint4*)&t16[8] = w1;
#pragma unroll
    for (int e = 0; e < 16; ++e) sB[buf][(b_n + e) * STR + b_k] = t16[e];
  };

  stage(0, 0);
#ifdef USE_ASYNC_LDS
  __builtin_amdgcn_s_wait_asynccnt(0);
#endif
  __syncthreads();

  int cur = 0;
  for (int k0 = 0; k0 < K; k0 += 32) {
    if (k0 + 32 < K) stage(k0 + 32, cur ^ 1);

    FragBF a0, a1, bf[4];
#pragma unroll
    for (int j = 0; j < 8; ++j) {
      int kk = (j >> 2) * 16 + g * 8 + ((j & 3) << 1);
      a0.u[j] = *(const unsigned*)&sA[cur][(wm + col) * STR + kk];
      a1.u[j] = *(const unsigned*)&sA[cur][(wm + 16 + col) * STR + kk];
    }
#pragma unroll
    for (int c = 0; c < 4; ++c)
#pragma unroll
      for (int j = 0; j < 8; ++j) {
        int kk = (j >> 2) * 16 + g * 8 + ((j & 3) << 1);
        bf[c].u[j] = *(const unsigned*)&sB[cur][(wn + c * 16 + col) * STR + kk];
      }
#pragma unroll
    for (int c = 0; c < 4; ++c) {
      acc[c]     = WMMA_BF16(a0.v, bf[c].v, acc[c]);
      acc[4 + c] = WMMA_BF16(a1.v, bf[c].v, acc[4 + c]);
    }

#ifdef USE_ASYNC_LDS
    __builtin_amdgcn_s_wait_asynccnt(0);
#endif
    __syncthreads();
    cur ^= 1;
  }

  // Epilogue (C layout: lanes 0-15 rows r, lanes 16-31 rows r+8; col = lane&15)
#pragma unroll
  for (int mi = 0; mi < 2; ++mi)
#pragma unroll
    for (int c = 0; c < 4; ++c) {
      v8f a = acc[mi * 4 + c];
#pragma unroll
      for (int r = 0; r < 8; ++r) {
        int m = m0 + wm + mi * 16 + r + 8 * g;
        int n = n0 + wn + c * 16 + col;
        if (m < M) {
          float v = a[r] * scale;
          if (bias) v += bias[n];
          if (act == 1) v = gelu_f(v);
          if (resid) v += resid[(size_t)m * N + n];
          if (Cf) Cf[(size_t)m * N + n] = v;
          if (Cb) Cb[(size_t)m * N + n] = f2bf(v);
        }
      }
    }
}

// ---------------------------------------------------------------------------
// Flash cross-attention. 128 threads = 4 waves sharing one (b,h); each wave
// owns a 16-query-row tile. K/V tiles (32 kv x 64 hd) are staged ONCE per
// block into double-buffered LDS: K via async global->LDS (coalesced b128),
// V transposed [hd][kv] so P@V fragments are packed b32 LDS reads.
// scores = q@k^T (WMMA), online softmax, o += P@v (WMMA).
// ---------------------------------------------------------------------------
__device__ __forceinline__ bool keep_kv(int qi, int kv) {
  if (kv >= KV_) return false;
  if (kv == 0) return true;              // temb column always visible
  if (kv <= S_) return (kv - 1) > qi;    // l2r keys: strictly future
  return (kv - S_ - 1) < qi;             // r2l keys: strictly past
}

__global__ __launch_bounds__(128) void attn_kernel(
    const unsigned short* __restrict__ qb, const unsigned short* __restrict__ kb,
    const unsigned short* __restrict__ vb, unsigned short* __restrict__ ob) {
  constexpr int KSTR = 72;  // 144B rows: 16B-aligned staging, conflict-free reads
  constexpr int VSTR = 40;  // 80B rows: conflict-free packed pair reads
  __shared__ __align__(16) unsigned short sK[2][32 * KSTR];
  __shared__ __align__(16) unsigned short sV[2][64 * VSTR];  // transposed [hd][kv]
  __shared__ __align__(8)  unsigned short sP[4][16 * 32];

  const int tid = threadIdx.x;
  const int wave = tid >> 5, lane = tid & 31;
  const int b = blockIdx.y / H_, h = blockIdx.y % H_;
  const int qbase = (blockIdx.x * 4 + wave) * 16;
  const int col = lane & 15, g = lane >> 4;

  const int s_r = tid >> 2;        // 0..31 : kv row in tile
  const int s_c = (tid & 3) * 16;  // 0..48 : hd col base (16 halves per thread)

  auto stageKV = [&](int t, int buf) {
    int krow = t * 32 + s_r;
    if (krow >= KV_) krow = KV_ - 1;  // clamp; masked columns never contribute
    const size_t rbase = (size_t)(b * KV_ + krow) * E_ + h * HD_ + s_c;
    // K tile: natural [kv][hd], async coalesced
    const unsigned short* kp = &kb[rbase];
#ifdef USE_ASYNC_LDS
    g2l_async16(kp, &sK[buf][s_r * KSTR + s_c]);
    g2l_async16(kp + 8, &sK[buf][s_r * KSTR + s_c + 8]);
#else
    *(uint4*)&sK[buf][s_r * KSTR + s_c] = *(const uint4*)kp;
    *(uint4*)&sK[buf][s_r * KSTR + s_c + 8] = *(const uint4*)(kp + 8);
#endif
    // V tile: transposed [hd][kv] via registers
    const unsigned short* vp = &vb[rbase];
    uint4 v0 = *(const uint4*)vp;
    uint4 v1 = *(const uint4*)(vp + 8);
    unsigned short tv[16];
    *(uint4*)&tv[0] = v0;
    *(uint4*)&tv[8] = v1;
#pragma unroll
    for (int e = 0; e < 16; ++e) sV[buf][(s_c + e) * VSTR + s_r] = tv[e];
  };

  // Q fragments: 16 x 64 split into two 16x32 A-operands (1/sqrt(HD) folded in)
  FragBF qf[2];
#pragma unroll
  for (int ks = 0; ks < 2; ++ks)
#pragma unroll
    for (int j = 0; j < 8; ++j) {
      int k = ks * 32 + (j >> 2) * 16 + g * 8 + ((j & 3) << 1);
      qf[ks].u[j] = *(const unsigned*)&qb[(size_t)(b * S_ + qbase + col) * E_ + h * HD_ + k];
    }

  float mrow[8], lrow[8];
  v8f o[4];
#pragma unroll
  for (int r = 0; r < 8; ++r) { mrow[r] = -3.0e38f; lrow[r] = 0.f; }
#pragma unroll
  for (int c = 0; c < 4; ++c) o[c] = (v8f){0.f, 0.f, 0.f, 0.f, 0.f, 0.f, 0.f, 0.f};

  const int ntiles = (KV_ + 31) / 32;
  stageKV(0, 0);
#ifdef USE_ASYNC_LDS
  __builtin_amdgcn_s_wait_asynccnt(0);
#endif
  __syncthreads();

  int cur = 0;
  for (int t = 0; t < ntiles; ++t) {
    if (t + 1 < ntiles) stageKV(t + 1, cur ^ 1);
    const int kv0 = t * 32;

    // scores: two 16-col sub-tiles, K reduction over HD=64 in two 32-steps
    v8f s0 = (v8f){0.f, 0.f, 0.f, 0.f, 0.f, 0.f, 0.f, 0.f};
    v8f s1 = (v8f){0.f, 0.f, 0.f, 0.f, 0.f, 0.f, 0.f, 0.f};
#pragma unroll
    for (int nt = 0; nt < 2; ++nt)
#pragma unroll
      for (int ks = 0; ks < 2; ++ks) {
        FragBF kf;
#pragma unroll
        for (int j = 0; j < 8; ++j) {
          int hd = ks * 32 + (j >> 2) * 16 + g * 8 + ((j & 3) << 1);
          kf.u[j] = *(const unsigned*)&sK[cur][(nt * 16 + col) * KSTR + hd];
        }
        if (nt == 0) s0 = WMMA_BF16(qf[ks].v, kf.v, s0);
        else         s1 = WMMA_BF16(qf[ks].v, kf.v, s1);
      }

    // mask + online softmax (row stats replicated across each 16-lane group)
#pragma unroll
    for (int r = 0; r < 8; ++r) {
      int qi = qbase + r + 8 * g;
      int c0 = kv0 + col, c1 = c0 + 16;
      float v0 = keep_kv(qi, c0) ? s0[r] : -1e30f;
      float v1 = keep_kv(qi, c1) ? s1[r] : -1e30f;
      float mx = fmaxf(v0, v1);
#pragma unroll
      for (int off = 1; off < 16; off <<= 1) mx = fmaxf(mx, __shfl_xor(mx, off, 16));
      float mnew = fmaxf(mrow[r], mx);
      float corr = __expf(mrow[r] - mnew);
      float p0 = __expf(v0 - mnew), p1 = __expf(v1 - mnew);
      float ps = p0 + p1;
#pragma unroll
      for (int off = 1; off < 16; off <<= 1) ps += __shfl_xor(ps, off, 16);
      lrow[r] = lrow[r] * corr + ps;
      mrow[r] = mnew;
#pragma unroll
      for (int c = 0; c < 4; ++c) o[c][r] *= corr;
      sP[wave][(r + 8 * g) * 32 + col] = f2bf(p0);
      sP[wave][(r + 8 * g) * 32 + 16 + col] = f2bf(p1);
    }
    __syncthreads();  // C-layout -> A-layout redistribution through LDS

    FragBF pf;
#pragma unroll
    for (int j = 0; j < 8; ++j) {
      int k = (j >> 2) * 16 + g * 8 + ((j & 3) << 1);
      pf.u[j] = *(const unsigned*)&sP[wave][col * 32 + k];
    }
#pragma unroll
    for (int c = 0; c < 4; ++c) {
      FragBF vf;
#pragma unroll
      for (int j = 0; j < 8; ++j) {
        int kk = (j >> 2) * 16 + g * 8 + ((j & 3) << 1);
        vf.u[j] = *(const unsigned*)&sV[cur][(c * 16 + col) * VSTR + kk];
      }
      o[c] = WMMA_BF16(pf.v, vf.v, o[c]);
    }

#ifdef USE_ASYNC_LDS
    __builtin_amdgcn_s_wait_asynccnt(0);
#endif
    __syncthreads();
    cur ^= 1;
  }

#pragma unroll
  for (int c = 0; c < 4; ++c)
#pragma unroll
    for (int r = 0; r < 8; ++r) {
      int qi = qbase + r + 8 * g;
      float val = o[c][r] / lrow[r];
      ob[(size_t)(b * S_ + qi) * E_ + h * HD_ + c * 16 + col] = f2bf(val);
    }
}

// ---------------------------------------------------------------------------
// x = film_a * LayerNorm(x + z) + film_b   (in-place on x, also bf16 copy)
// ---------------------------------------------------------------------------
__global__ __launch_bounds__(256) void ln_film_kernel(
    float* __restrict__ x, const float* __restrict__ z,
    const float* __restrict__ g, const float* __restrict__ bt,
    const float* __restrict__ film, unsigned short* __restrict__ xb) {
  int rowid = blockIdx.x;
  int b = rowid / S_;
  int tid = threadIdx.x;
  float* px = x + (size_t)rowid * E2_;
  const float* pz = z + (size_t)rowid * E2_;

  float y[E2_ / 256];
  float s1 = 0.f, q1 = 0.f;
#pragma unroll
  for (int i = 0; i < E2_ / 256; ++i) {
    int e = tid + i * 256;
    float v = px[e] + pz[e];
    y[i] = v; s1 += v; q1 += v * v;
  }
  __shared__ float sh[512];
  sh[tid] = s1; sh[256 + tid] = q1;
  __syncthreads();
  for (int o = 128; o > 0; o >>= 1) {
    if (tid < o) { sh[tid] += sh[tid + o]; sh[256 + tid] += sh[256 + tid + o]; }
    __syncthreads();
  }
  float m = sh[0] / (float)E2_;
  float var = sh[256] / (float)E2_ - m * m;
  float rstd = rsqrtf(var + 1e-5f);
#pragma unroll
  for (int i = 0; i < E2_ / 256; ++i) {
    int e = tid + i * 256;
    float yn = (y[i] - m) * rstd * g[e] + bt[e];
    float a = film[(size_t)b * E4_ + e];
    float bb = film[(size_t)b * E4_ + E2_ + e];
    float out = a * yn + bb;
    px[e] = out;
    xb[(size_t)rowid * E2_ + e] = f2bf(out);
  }
}

// ---------------------------------------------------------------------------
extern "C" void kernel_launch(void* const* d_in, const int* in_sizes, int n_in,
                              void* d_out, int out_size, void* d_ws, size_t ws_size,
                              hipStream_t stream) {
  (void)in_sizes; (void)n_in; (void)out_size; (void)ws_size;

  const float* l2r     = (const float*)d_in[0];
  const float* r2l     = (const float*)d_in[1];
  const float* temb    = (const float*)d_in[2];
  const float* Wq      = (const float*)d_in[3];
  const float* Wk      = (const float*)d_in[4];
  const float* bk      = (const float*)d_in[5];
  const float* Wv      = (const float*)d_in[6];
  const float* bv      = (const float*)d_in[7];
  const float* Wo      = (const float*)d_in[8];
  const float* bo      = (const float*)d_in[9];
  const float* ln1_g   = (const float*)d_in[10];
  const float* ln1_b   = (const float*)d_in[11];
  const float* ln2_g   = (const float*)d_in[12];
  const float* ln2_b   = (const float*)d_in[13];
  const float* mlp_w1  = (const float*)d_in[14];
  const float* mlp_b1  = (const float*)d_in[15];
  const float* mlp_w2  = (const float*)d_in[16];
  const float* mlp_b2  = (const float*)d_in[17];
  const float* in_w    = (const float*)d_in[18];
  const float* in_b    = (const float*)d_in[19];
  const float* res_w1  = (const float*)d_in[20];
  const float* res_b1  = (const float*)d_in[21];
  const float* res_w2  = (const float*)d_in[22];
  const float* res_b2  = (const float*)d_in[23];
  const float* resln_g = (const float*)d_in[24];
  const float* resln_b = (const float*)d_in[25];
  const float* film_w  = (const float*)d_in[26];
  const float* film_b  = (const float*)d_in[27];
  const float* log_w   = (const float*)d_in[28];
  const float* log_b   = (const float*)d_in[29];

  char* p = (char*)d_ws;
  auto alloc = [&](size_t elems, size_t esz) -> void* {
    void* r = (void*)p;
    p += ((elems * esz + 255) & ~(size_t)255);
    return r;
  };

  const size_t EE = (size_t)E_ * E_;
  unsigned short* wq_b   = (unsigned short*)alloc(EE, 2);
  unsigned short* wk_b   = (unsigned short*)alloc(EE, 2);
  unsigned short* wv_b   = (unsigned short*)alloc(EE, 2);
  unsigned short* wo_b   = (unsigned short*)alloc(EE, 2);
  unsigned short* mw1_b  = (unsigned short*)alloc((size_t)E_ * MLP_, 2);
  unsigned short* mw2_b  = (unsigned short*)alloc((size_t)MLP_ * E4_, 2);
  unsigned short* inw_b  = (unsigned short*)alloc((size_t)E_ * E2_, 2);
  unsigned short* rw1_b  = (unsigned short*)alloc((size_t)L_ * E2_ * MLP_, 2);
  unsigned short* rw2_b  = (unsigned short*)alloc((size_t)L_ * MLP_ * E2_, 2);
  unsigned short* fw_b   = (unsigned short*)alloc((size_t)L_ * E4_ * E4_, 2);
  unsigned short* lw_b   = (unsigned short*)alloc((size_t)E2_ * SOUT_, 2);

  float*          inputs_f = (float*)alloc((size_t)B_ * S_ * E_, 4);
  unsigned short* qin_b    = (unsigned short*)alloc((size_t)B_ * S_ * E_, 2);
  unsigned short* allemb_b = (unsigned short*)alloc((size_t)B_ * KV_ * E_, 2);
  unsigned short* tembb    = (unsigned short*)alloc((size_t)B_ * E_, 2);
  unsigned short* qbuf     = (unsigned short*)alloc((size_t)B_ * S_ * E_, 2);
  unsigned short* kbuf     = (unsigned short*)alloc((size_t)B_ * KV_ * E_, 2);
  unsigned short* vbuf     = (unsigned short*)alloc((size_t)B_ * KV_ * E_, 2);
  unsigned short* attnb    = (unsigned short*)alloc((size_t)B_ * S_ * E_, 2);
  unsigned short* xres_b   = (unsigned short*)alloc((size_t)B_ * S_ * E_, 2);
  unsigned short* tg_b     = (unsigned short*)alloc((size_t)B_ * MLP_, 2);
  unsigned short* t_b      = (unsigned short*)alloc((size_t)B_ * E4_, 2);
  float*          x0_f     = (float*)alloc((size_t)B_ * S_ * E2_, 4);
  unsigned short* x0_b     = (unsigned short*)alloc((size_t)B_ * S_ * E2_, 2);
  unsigned short* h_b      = (unsigned short*)alloc((size_t)B_ * S_ * MLP_, 2);
  float*          z_f      = (float*)alloc((size_t)B_ * S_ * E2_, 4);
  float*          film_f   = (float*)alloc((size_t)B_ * E4_, 4);

  auto conv = [&](const float* src, unsigned short* dst, size_t n) {
    int grid = (int)((n + 255) / 256);
    if (grid > 4096) grid = 4096;
    f2bf_kernel<<<grid, 256, 0, stream>>>(src, dst, (int)n);
  };
  conv(Wq, wq_b, EE);
  conv(Wk, wk_b, EE);
  conv(Wv, wv_b, EE);
  conv(Wo, wo_b, EE);
  conv(mlp_w1, mw1_b, (size_t)E_ * MLP_);
  conv(mlp_w2, mw2_b, (size_t)MLP_ * E4_);
  conv(in_w, inw_b, (size_t)E_ * E2_);
  conv(res_w1, rw1_b, (size_t)L_ * E2_ * MLP_);
  conv(res_w2, rw2_b, (size_t)L_ * MLP_ * E2_);
  conv(film_w, fw_b, (size_t)L_ * E4_ * E4_);
  conv(log_w, lw_b, (size_t)E2_ * SOUT_);

  prep_kernel<<<B_ * S_, 256, 0, stream>>>(l2r, r2l, ln1_g, ln1_b, ln2_g, ln2_b,
                                           inputs_f, qin_b, allemb_b);
  temb_prep_kernel<<<B_, 256, 0, stream>>>(temb, tembb, allemb_b);

  auto gemm = [&](const unsigned short* A, const unsigned short* Wt, const float* bias,
                  const float* resid, float* Cf, unsigned short* Cb,
                  int M, int N, int K, float scale, int act) {
    dim3 grd(N / 128, (M + 127) / 128);
    gemm_bf16_kernel<<<grd, 256, 0, stream>>>(A, Wt, bias, resid, Cf, Cb, M, N, K, scale, act);
  };

  const int BS = B_ * S_;    // 4096
  const int BKV = B_ * KV_;  // 8196

  // q = ((ln1+ln2) @ Wq) / sqrt(HD)   (scale folded here)
  gemm(qin_b, wq_b, nullptr, nullptr, nullptr, qbuf, BS, E_, E_, 0.125f, 0);
  // k = all_embed @ Wk + bk ; v = all_embed @ Wv + bv
  gemm(allemb_b, wk_b, bk, nullptr, nullptr, kbuf, BKV, E_, E_, 1.f, 0);
  gemm(allemb_b, wv_b, bv, nullptr, nullptr, vbuf, BKV, E_, E_, 1.f, 0);

  attn_kernel<<<dim3(S_ / 64, B_ * H_), 128, 0, stream>>>(qbuf, kbuf, vbuf, attnb);

  // x = attn @ Wo + bo + inputs
  gemm(attnb, wo_b, bo, inputs_f, nullptr, xres_b, BS, E_, E_, 1.f, 0);

  // t = gelu(temb @ mlp_w1 + b1) @ mlp_w2 + b2
  gemm(tembb, mw1_b, mlp_b1, nullptr, nullptr, tg_b, B_, MLP_, E_, 1.f, 1);
  gemm(tg_b, mw2_b, mlp_b2, nullptr, nullptr, t_b, B_, E4_, MLP_, 1.f, 0);

  // x = x @ in_w + in_b
  gemm(xres_b, inw_b, in_b, nullptr, x0_f, x0_b, BS, E2_, E_, 1.f, 0);

  for (int i = 0; i < L_; ++i) {
    // film = t @ film_w[i] + film_b[i]
    gemm(t_b, fw_b + (size_t)i * E4_ * E4_, film_b + (size_t)i * E4_, nullptr,
         film_f, nullptr, B_, E4_, E4_, 1.f, 0);
    // h = gelu(x @ res_w1[i] + res_b1[i])
    gemm(x0_b, rw1_b + (size_t)i * E2_ * MLP_, res_b1 + (size_t)i * MLP_, nullptr,
         nullptr, h_b, BS, MLP_, E2_, 1.f, 1);
    // z = h @ res_w2[i] + res_b2[i]
    gemm(h_b, rw2_b + (size_t)i * MLP_ * E2_, res_b2 + (size_t)i * E2_, nullptr,
         z_f, nullptr, BS, E2_, MLP_, 1.f, 0);
    // x = a * LN(x+z) + b  (FiLM)
    ln_film_kernel<<<BS, 256, 0, stream>>>(x0_f, z_f, resln_g + (size_t)i * E2_,
                                           resln_b + (size_t)i * E2_, film_f, x0_b);
  }

  // out = x @ log_w + log_b
  gemm(x0_b, lw_b, log_b, nullptr, (float*)d_out, nullptr, BS, SOUT_, E2_, 1.f, 0);
}
